// RPN_15479062135172
// MI455X (gfx1250) — matrix operand order repaired and checked by reading, not compiled, not run on hardware
//
#include <hip/hip_runtime.h>
#include <hip/hip_bf16.h>
#include <float.h>

// ---------------------------------------------------------------------------
// CDNA5 (gfx1250) RPN proposal pipeline.
// Heavy math (3x3 conv + fused 1x1 heads) runs on v_wmma_f32_16x16x32_bf16.
// Conv: implicit GEMM M=3800 N=512 K=9216, 64x64 block tile, double-buffered
// LDS staging (1 barrier / K-step), 2 WMMA per wave per step.
// ---------------------------------------------------------------------------

typedef __attribute__((ext_vector_type(16))) __bf16 v16bf;
typedef __attribute__((ext_vector_type(8)))  float  v8f;

#define FH 50
#define FW 76
#define M_SP   3800          // 50*76 spatial positions
#define M_PAD  3840          // padded to 60 * 64
#define CIN    1024
#define CHID   512
#define NCOMB  64            // 18 cls + 36 reg + 10 pad
#define NPROP  34200         // 3800 * 9 anchors
#define PRE_TOPN  4000
#define POST_TOPN 2000
#define NEG_INF_F (-1000000000.0f)
#define MASK_W 128           // uint32 words per NMS mask row (125 used)

// ---- workspace layout (bytes) ----
static constexpr size_t HID_OFF   = 0;               // 3840*512*2  = 3932160
static constexpr size_t WCOMB_OFF = 3932160;         // 512*64*2    = 65536
static constexpr size_t OUTC_OFF  = 3997696;         // 3840*64*4   = 983040
static constexpr size_t PROP_OFF  = 4980736;         // 34200*4*4   = 547200
static constexpr size_t MS_OFF    = 5527936;         // 34200*4     = 136800
static constexpr size_t HIST_OFF  = 5664768;         // 2048*4      = 8192
static constexpr size_t META_OFF  = 5672960;         // [0]=thrkey [1]=count
static constexpr size_t CAND_OFF  = 5673216;         // 34200*4     = 136800
static constexpr size_t TOPB_OFF  = 5810176;         // 4000*4*4    = 64000
static constexpr size_t TOPS_OFF  = 5874176;         // 4000*4      = 16000
static constexpr size_t MASK_OFF  = 5890304;         // 4000*128*4  = 2048000
// total ~7.94 MB

// generate_anchors(16,(.5,1,2),(8,16,32)) widths/heights (exact, precomputed)
__device__ const float ANCH_W[9] = {184.f,368.f,736.f,128.f,256.f,512.f, 88.f,176.f,352.f};
__device__ const float ANCH_H[9] = { 96.f,192.f,384.f,128.f,256.f,512.f,176.f,352.f,704.f};

// ---------------------------------------------------------------------------
// 0) zero d_out (10000 floats) + histogram (2048 bins)
// ---------------------------------------------------------------------------
__global__ void zero_kernel(float* __restrict__ out, unsigned* __restrict__ hist) {
  int i = blockIdx.x * 256 + threadIdx.x;
  if (i < 10000) out[i] = 0.0f;
  else if (i < 10000 + 2048) hist[i - 10000] = 0u;
}

// ---------------------------------------------------------------------------
// 1) build combined 1x1 head weights: wcomb[k=cin][n] (bf16), n: 0-17 cls,
//    18-53 reg, 54-63 zero pad
// ---------------------------------------------------------------------------
__global__ void prep_wcomb(const float* __restrict__ wcls,
                           const float* __restrict__ wreg,
                           __bf16* __restrict__ wcomb) {
  int idx = blockIdx.x * 256 + threadIdx.x;       // 512*64
  if (idx >= CHID * NCOMB) return;
  int k = idx >> 6;
  int n = idx & 63;
  float v = 0.0f;
  if (n < 18)      v = wcls[n * CHID + k];
  else if (n < 54) v = wreg[(n - 18) * CHID + k];
  wcomb[k * NCOMB + n] = (__bf16)v;
}

// ---------------------------------------------------------------------------
// 2) 3x3 conv as implicit GEMM. 288 K-steps (9 taps x 32 channel-blocks).
//    block = 256 thr = 8 waves; block tile 64(M) x 64(N); wave owns 16x32
//    (two 16x16 accumulators sharing one A fragment). Double-buffered LDS,
//    one barrier per K-step. grid (60, 8). Epilogue: +bias, ReLU, bf16 store.
// ---------------------------------------------------------------------------
__global__ __launch_bounds__(256) void conv3x3_wmma(
    const float* __restrict__ feats, const float* __restrict__ wconv,
    const float* __restrict__ bconv, __bf16* __restrict__ hid) {
  __shared__ __bf16 As[2][64][40];   // [buf][m][k], padded row stride
  __shared__ __bf16 Bt[2][64][40];   // [buf][n][k]
  const int tid  = threadIdx.x;
  const int lane = tid & 31;
  const int wave = tid >> 5;
  const int wm   = wave >> 1;        // 0..3  (M sub-tile)
  const int wn   = wave & 1;         // 0..1  (N half)
  const int m0   = blockIdx.x * 64;
  const int n0   = blockIdx.y * 64;

  // per-thread staging geometry
  const int ml = tid & 63;           // A row this thread stages
  const int gm = m0 + ml;
  const int ay = gm / FW, ax = gm - ay * FW;
  const int cA = tid >> 6;           // 0..3  A column group
  const int cB = tid & 31;           // B k-column
  const int nB = tid >> 5;           // 0..7  B row group

  v8f acc0 = {}, acc1 = {};

  auto stage = [&](int s, int buf) {
    const int tap = s >> 5;                  // 0..8
    const int cb  = (s & 31) << 5;           // channel block base
    const int ky = tap / 3 - 1, kx = tap % 3 - 1;
    const int yy = ay + ky, xx = ax + kx;
    const bool mok = (gm < M_SP) && (yy >= 0) && (yy < FH) && (xx >= 0) && (xx < FW);
    const int abase = mok ? (yy * FW + xx) : 0;   // clamped-safe address
    __builtin_prefetch(&feats[(cb + 32) * M_SP + abase], 0, 3);
#pragma unroll
    for (int r = 0; r < 8; ++r) {            // A: 64m x 32c, coalesced along m
      const int cl = r * 4 + cA;
      float v = feats[(cb + cl) * M_SP + abase];
      As[buf][ml][cl] = (__bf16)(mok ? v : 0.0f);
    }
#pragma unroll
    for (int r = 0; r < 8; ++r) {            // B: Bt[n][c] from OIHW weights
      const int nl = r * 8 + nB;
      float v = wconv[((n0 + nl) * CIN + cb + cB) * 9 + tap];
      Bt[buf][nl][cB] = (__bf16)v;
    }
  };

  stage(0, 0);
  __syncthreads();

  const int h  = lane >> 4;                  // half-wave
  const int ar = wm * 16 + (lane & 15);      // A row (M)
  const int bc = wn * 32 + (lane & 15);      // B col (N), tiles at bc, bc+16

  for (int s = 0; s < 288; ++s) {
    const int buf = s & 1;
    if (s + 1 < 288) stage(s + 1, buf ^ 1);  // overlap next-stage VMEM w/ WMMA
    v16bf a, b0, b1;
#pragma unroll
    for (int i = 0; i < 16; ++i) {
      // A 16x32: K = h*16 + 0..15
      a[i]  = As[buf][ar][h * 16 + i];
      // B 32x16: K = h*16 + 0..15
      b0[i] = Bt[buf][bc][h * 16 + i];
      b1[i] = Bt[buf][bc + 16][h * 16 + i];
    }
    acc0 = __builtin_amdgcn_wmma_f32_16x16x32_bf16(
        false, a, false, b0, (short)0, acc0, false, false);
    acc1 = __builtin_amdgcn_wmma_f32_16x16x32_bf16(
        false, a, false, b1, (short)0, acc1, false, false);
    __syncthreads();
  }

  // epilogue: bias + ReLU, store bf16 hid[m][n] (NHWC)
  const int na = n0 + wn * 32 + (lane & 15);
  const int nb = na + 16;
  const float bias0 = bconv[na];
  const float bias1 = bconv[nb];
#pragma unroll
  for (int v = 0; v < 8; ++v) {
    const int m = m0 + wm * 16 + h * 8 + v;  // C/D: M = vgpr + 8*half
    float v0 = acc0[v] + bias0;
    float v1 = acc1[v] + bias1;
    hid[m * CHID + na] = (__bf16)(v0 > 0.0f ? v0 : 0.0f);
    hid[m * CHID + nb] = (__bf16)(v1 > 0.0f ? v1 : 0.0f);
  }
}

// ---------------------------------------------------------------------------
// 3) fused 1x1 heads: outc[3840][64] = hid(3840x512 bf16) x wcomb(512x64 bf16)
//    64x64 block tile, 2 accumulators/wave, 16 K-steps. grid (60, 1).
// ---------------------------------------------------------------------------
__global__ __launch_bounds__(256) void head_wmma(
    const __bf16* __restrict__ hid, const __bf16* __restrict__ wcomb,
    float* __restrict__ outc) {
  __shared__ __bf16 As[64][40];
  __shared__ __bf16 Bt[64][40];
  const int tid  = threadIdx.x;
  const int lane = tid & 31;
  const int wave = tid >> 5;
  const int wm   = wave >> 1;
  const int wn   = wave & 1;
  const int m0   = blockIdx.x * 64;

  v8f acc0 = {}, acc1 = {};
  const int h  = lane >> 4;
  const int ar = wm * 16 + (lane & 15);
  const int bc = wn * 32 + (lane & 15);

  for (int k0 = 0; k0 < CHID; k0 += 32) {
    // stage A: coalesced along k (contiguous in hid row)
    {
      const int kl = tid & 31;
#pragma unroll
      for (int r = 0; r < 8; ++r) {
        const int ml = r * 8 + (tid >> 5);
        As[ml][kl] = hid[(m0 + ml) * CHID + k0 + kl];
      }
    }
    // stage B: Bt[n][k], coalesced along n (wcomb row is 64 wide)
    {
      const int nl = tid & 63;
#pragma unroll
      for (int r = 0; r < 8; ++r) {
        const int kl = r * 4 + (tid >> 6);
        Bt[nl][kl] = wcomb[(k0 + kl) * NCOMB + nl];
      }
    }
    __syncthreads();
    v16bf a, b0, b1;
#pragma unroll
    for (int i = 0; i < 16; ++i) {
      a[i]  = As[ar][h * 16 + i];
      b0[i] = Bt[bc][h * 16 + i];
      b1[i] = Bt[bc + 16][h * 16 + i];
    }
    acc0 = __builtin_amdgcn_wmma_f32_16x16x32_bf16(
        false, a, false, b0, (short)0, acc0, false, false);
    acc1 = __builtin_amdgcn_wmma_f32_16x16x32_bf16(
        false, a, false, b1, (short)0, acc1, false, false);
    __syncthreads();
  }
  const int na = wn * 32 + (lane & 15);
  const int nb = na + 16;
#pragma unroll
  for (int v = 0; v < 8; ++v) {
    const int m = m0 + wm * 16 + h * 8 + v;
    outc[m * NCOMB + na] = acc0[v];
    outc[m * NCOMB + nb] = acc1[v];
  }
}

// ---------------------------------------------------------------------------
// 4) anchors + bbox_transform_inv + clip + min-size mask
// ---------------------------------------------------------------------------
__global__ void proposals_kernel(const float* __restrict__ outc,
                                 const float* __restrict__ bcls,
                                 const float* __restrict__ breg,
                                 float* __restrict__ props,
                                 float* __restrict__ ms) {
  const int p = blockIdx.x * 256 + threadIdx.x;
  if (p >= NPROP) return;
  const int m = p / 9, a = p - m * 9;
  const int y = m / FW, x = m - y * FW;
  const float sx = (float)(x * 16), sy = (float)(y * 16);

  const float score = outc[m * NCOMB + 9 + a] + bcls[9 + a];
  const float d0 = outc[m * NCOMB + 18 + 4 * a + 0] + breg[4 * a + 0];
  const float d1 = outc[m * NCOMB + 18 + 4 * a + 1] + breg[4 * a + 1];
  const float d2 = outc[m * NCOMB + 18 + 4 * a + 2] + breg[4 * a + 2];
  const float d3 = outc[m * NCOMB + 18 + 4 * a + 3] + breg[4 * a + 3];

  const float aw = ANCH_W[a], ah = ANCH_H[a];
  // anchor center = (xc + 0.5 + shift) = 8 + shift for base_size 16
  const float pcx = d0 * aw + sx + 8.0f;
  const float pcy = d1 * ah + sy + 8.0f;
  const float pw  = expf(d2) * aw;
  const float ph  = expf(d3) * ah;
  float x1 = pcx - 0.5f * pw, y1 = pcy - 0.5f * ph;
  float x2 = pcx + 0.5f * pw, y2 = pcy + 0.5f * ph;
  x1 = fminf(fmaxf(x1, 0.0f), 1215.0f);
  x2 = fminf(fmaxf(x2, 0.0f), 1215.0f);
  y1 = fminf(fmaxf(y1, 0.0f), 799.0f);
  y2 = fminf(fmaxf(y2, 0.0f), 799.0f);
  props[p * 4 + 0] = x1; props[p * 4 + 1] = y1;
  props[p * 4 + 2] = x2; props[p * 4 + 3] = y2;
  const bool valid = ((x2 - x1 + 1.0f) >= 16.0f) && ((y2 - y1 + 1.0f) >= 16.0f);
  ms[p] = valid ? score : NEG_INF_F;
}

// ---------------------------------------------------------------------------
// 5-8) exact top-4000 (argsort-stable) via radix histogram + rank count
// ---------------------------------------------------------------------------
__device__ inline unsigned fkey(float f) {  // order-preserving float->uint
  unsigned u = __float_as_uint(f);
  return (u & 0x80000000u) ? ~u : (u | 0x80000000u);
}

__global__ void hist_kernel(const float* __restrict__ ms, unsigned* __restrict__ hist) {
  const int p = blockIdx.x * 256 + threadIdx.x;
  if (p < NPROP) atomicAdd(&hist[fkey(ms[p]) >> 21], 1u);
}

__global__ void select_kernel(const unsigned* __restrict__ hist,
                              unsigned* __restrict__ meta) {
  if (threadIdx.x == 0 && blockIdx.x == 0) {
    unsigned accum = 0;
    int b = 2047;
    for (; b >= 0; --b) { accum += hist[b]; if (accum >= PRE_TOPN) break; }
    if (b < 0) b = 0;
    meta[0] = ((unsigned)b) << 21;  // min key admitted
    meta[1] = 0u;                   // candidate counter
  }
}

__global__ void compact_kernel(const float* __restrict__ ms,
                               unsigned* __restrict__ meta,
                               int* __restrict__ cand) {
  const int p = blockIdx.x * 256 + threadIdx.x;
  if (p < NPROP && fkey(ms[p]) >= meta[0]) {
    unsigned pos = atomicAdd(&meta[1], 1u);
    cand[pos] = p;
  }
}

__global__ __launch_bounds__(256) void rank_kernel(
    const float* __restrict__ ms, const float* __restrict__ props,
    const unsigned* __restrict__ meta, const int* __restrict__ cand,
    float* __restrict__ topb, float* __restrict__ tops) {
  __shared__ float ss[256];
  __shared__ int   sj[256];
  const int nc  = (int)meta[1];
  const int c   = blockIdx.x * 256 + threadIdx.x;
  const bool live = c < nc;
  const int   i  = live ? cand[c] : 0;
  const float si = live ? ms[i] : 0.0f;
  int rank = 0;
  for (int base = 0; base < nc; base += 256) {
    const int j = base + threadIdx.x;
    if (j < nc) { int jj = cand[j]; sj[threadIdx.x] = jj; ss[threadIdx.x] = ms[jj]; }
    else        { sj[threadIdx.x] = 0x7fffffff; ss[threadIdx.x] = -FLT_MAX; }
    __syncthreads();
    const int lim = (nc - base < 256) ? (nc - base) : 256;
    for (int t = 0; t < lim; ++t) {
      const float sjv = ss[t];
      const int   jj  = sj[t];
      rank += ((sjv > si) || ((sjv == si) && (jj < i))) ? 1 : 0;
    }
    __syncthreads();
  }
  if (live && rank < PRE_TOPN) {
    topb[rank * 4 + 0] = props[i * 4 + 0];
    topb[rank * 4 + 1] = props[i * 4 + 1];
    topb[rank * 4 + 2] = props[i * 4 + 2];
    topb[rank * 4 + 3] = props[i * 4 + 3];
    tops[rank] = si;
  }
}

// ---------------------------------------------------------------------------
// 9) IoU > 0.7 bitmask rows (4000 x 125 words), then
// 10) single-block sequential greedy sweep + scatter first 2000 kept
// ---------------------------------------------------------------------------
__global__ __launch_bounds__(128) void iou_kernel(const float* __restrict__ topb,
                                                  unsigned* __restrict__ mask) {
  const int i = blockIdx.x;
  const int w = threadIdx.x;
  const float x1 = topb[i * 4 + 0], y1 = topb[i * 4 + 1];
  const float x2 = topb[i * 4 + 2], y2 = topb[i * 4 + 3];
  const float ai = (x2 - x1 + 1.0f) * (y2 - y1 + 1.0f);
  unsigned bits = 0u;
  if (w < 125) {
    for (int b = 0; b < 32; ++b) {
      const int j = w * 32 + b;
      if (j >= PRE_TOPN) break;
      const float bx1 = topb[j * 4 + 0], by1 = topb[j * 4 + 1];
      const float bx2 = topb[j * 4 + 2], by2 = topb[j * 4 + 3];
      const float xx1 = fmaxf(x1, bx1), yy1 = fmaxf(y1, by1);
      const float xx2 = fminf(x2, bx2), yy2 = fminf(y2, by2);
      const float inter = fmaxf(0.0f, xx2 - xx1 + 1.0f) * fmaxf(0.0f, yy2 - yy1 + 1.0f);
      const float aj = (bx2 - bx1 + 1.0f) * (by2 - by1 + 1.0f);
      const float iou = inter / (ai + aj - inter);
      bits |= (iou > 0.7f) ? (1u << b) : 0u;
    }
  }
  if (w < MASK_W) mask[i * MASK_W + w] = bits;
}

__global__ __launch_bounds__(128) void nms_kernel(const float* __restrict__ topb,
                                                  const float* __restrict__ tops,
                                                  const unsigned* __restrict__ mask,
                                                  float* __restrict__ out) {
  __shared__ unsigned removed[128];
  __shared__ int kflag;
  const int tid = threadIdx.x;
  removed[tid] = 0u;
  int cnt = 0;  // only tid 0 uses
  __syncthreads();
  for (int i = 0; i < PRE_TOPN; ++i) {
    if (tid == 0) {
      const bool k = (tops[i] > NEG_INF_F * 0.5f) &&
                     !((removed[i >> 5] >> (i & 31)) & 1u);
      kflag = k ? 1 : 0;
    }
    __syncthreads();
    if (kflag) {
      removed[tid] |= mask[i * MASK_W + tid];
      if (tid == 0 && cnt < POST_TOPN) {
        out[cnt * 4 + 0] = topb[i * 4 + 0];
        out[cnt * 4 + 1] = topb[i * 4 + 1];
        out[cnt * 4 + 2] = topb[i * 4 + 2];
        out[cnt * 4 + 3] = topb[i * 4 + 3];
        out[4 * POST_TOPN + cnt] = tops[i];
        ++cnt;
      }
    }
    __syncthreads();
  }
}

// ---------------------------------------------------------------------------
extern "C" void kernel_launch(void* const* d_in, const int* in_sizes, int n_in,
                              void* d_out, int out_size, void* d_ws, size_t ws_size,
                              hipStream_t stream) {
  const float* feats = (const float*)d_in[1];
  const float* wconv = (const float*)d_in[2];
  const float* bconv = (const float*)d_in[3];
  const float* wcls  = (const float*)d_in[4];
  const float* bcls  = (const float*)d_in[5];
  const float* wreg  = (const float*)d_in[6];
  const float* breg  = (const float*)d_in[7];
  float* out = (float*)d_out;
  char*  ws  = (char*)d_ws;

  __bf16*   hid   = (__bf16*)(ws + HID_OFF);
  __bf16*   wcomb = (__bf16*)(ws + WCOMB_OFF);
  float*    outc  = (float*)(ws + OUTC_OFF);
  float*    props = (float*)(ws + PROP_OFF);
  float*    ms    = (float*)(ws + MS_OFF);
  unsigned* hist  = (unsigned*)(ws + HIST_OFF);
  unsigned* meta  = (unsigned*)(ws + META_OFF);
  int*      cand  = (int*)(ws + CAND_OFF);
  float*    topb  = (float*)(ws + TOPB_OFF);
  float*    tops  = (float*)(ws + TOPS_OFF);
  unsigned* mask  = (unsigned*)(ws + MASK_OFF);

  zero_kernel<<<48, 256, 0, stream>>>(out, hist);
  prep_wcomb<<<128, 256, 0, stream>>>(wcls, wreg, wcomb);
  conv3x3_wmma<<<dim3(M_PAD / 64, CHID / 64), 256, 0, stream>>>(feats, wconv, bconv, hid);
  head_wmma<<<dim3(M_PAD / 64, 1), 256, 0, stream>>>(hid, wcomb, outc);
  proposals_kernel<<<(NPROP + 255) / 256, 256, 0, stream>>>(outc, bcls, breg, props, ms);
  hist_kernel<<<(NPROP + 255) / 256, 256, 0, stream>>>(ms, hist);
  select_kernel<<<1, 32, 0, stream>>>(hist, meta);
  compact_kernel<<<(NPROP + 255) / 256, 256, 0, stream>>>(ms, meta, cand);
  rank_kernel<<<(NPROP + 255) / 256, 256, 0, stream>>>(ms, props, meta, cand, topb, tops);
  iou_kernel<<<PRE_TOPN, 128, 0, stream>>>(topb, mask);
  nms_kernel<<<1, 128, 0, stream>>>(topb, tops, mask, out);
}
